// KMeans_14121852469569
// MI455X (gfx1250) — matrix operand order repaired and checked by reading, not compiled, not run on hardware
//
#include <hip/hip_runtime.h>

// Problem dimensions (fixed by the harness' setup_inputs()).
#define NSs    1000
#define NQs    131072
#define Dd     1024
#define Cc     100
#define CP     112      // C padded to 7 WMMA n-tiles of 16
#define NIT    10
#define KSPLIT 128      // split-K blocks over queries in gemm2 (1024 queries each)

typedef unsigned short u16;
typedef __bf16 bf16;
typedef bf16  v16bf __attribute__((ext_vector_type(16)));
typedef float v8f   __attribute__((ext_vector_type(8)));

union BF16Frag { int4 q[2]; v16bf v; };   // 32 bytes: 16 bf16 elements
union Pack8    { u16 h[8]; int4 q; };     // 16 bytes: 8 bf16 elements

__device__ __forceinline__ u16 f2bf(float f) {
  unsigned u = __float_as_uint(f);
  unsigned r = u + 0x7FFFu + ((u >> 16) & 1u);   // round-to-nearest-even
  return (u16)(r >> 16);
}
__device__ __forceinline__ float bf2f(u16 b) {
  return __uint_as_float(((unsigned)b) << 16);
}

// ---------------------------------------------------------------- mean of support
__global__ void mean_k(const float* __restrict__ sx, float* __restrict__ mean) {
  int d = blockIdx.x * blockDim.x + threadIdx.x;
  if (d >= Dd) return;
  float s = 0.f;
  for (int i = 0; i < NSs; ++i) s += sx[(size_t)i * Dd + d];
  mean[d] = s * (1.0f / NSs);
}

// ---------------------------------------------------------------- zero helpers
__global__ void zero_sup_k(float* __restrict__ sup_sum, float* __restrict__ sup_cnt) {
  int i = blockIdx.x * 256 + threadIdx.x;
  if (i < CP * Dd) sup_sum[i] = 0.f;
  if (i < CP)      sup_cnt[i] = 0.f;
}
__global__ void zero_wqp_k(float* __restrict__ wqp) {
  size_t i = (size_t)blockIdx.x * 256 + threadIdx.x;
  if (i < (size_t)KSPLIT * CP * Dd) wqp[i] = 0.f;
}
__global__ void zero_iter_k(float* __restrict__ wsum, int* __restrict__ correct) {
  int i = threadIdx.x;
  if (i < CP) wsum[i] = 0.f;
  if (i == 0) *correct = 0;
}

// ---------------------------------------------------------------- support: normalize + class sums
__global__ __launch_bounds__(256)
void sup_k(const float* __restrict__ sx, const int* __restrict__ sy,
           const float* __restrict__ mean, float* __restrict__ sup_sum,
           float* __restrict__ sup_cnt) {
  int row = blockIdx.x, tid = threadIdx.x;
  __shared__ float red[256];
  float x[4]; float ss = 0.f;
#pragma unroll
  for (int k = 0; k < 4; ++k) {
    int d = k * 256 + tid;
    x[k] = sx[(size_t)row * Dd + d] - mean[d];
    ss += x[k] * x[k];
  }
  red[tid] = ss; __syncthreads();
  for (int s = 128; s > 0; s >>= 1) { if (tid < s) red[tid] += red[tid + s]; __syncthreads(); }
  float rn = rsqrtf(red[0]);
  int y = sy[row];
#pragma unroll
  for (int k = 0; k < 4; ++k) {
    int d = k * 256 + tid;
    atomicAdd(&sup_sum[(size_t)y * Dd + d], x[k] * rn);
  }
  if (tid == 0) atomicAdd(&sup_cnt[y], 1.0f);
}

// ---------------------------------------------------------------- query normalize -> bf16 row-major
__global__ __launch_bounds__(256)
void qnorm_k(const float* __restrict__ qx, const float* __restrict__ mean,
             u16* __restrict__ qn) {
  int row = blockIdx.x, tid = threadIdx.x;
  __shared__ float red[256];
  float x[4]; float ss = 0.f;
#pragma unroll
  for (int k = 0; k < 4; ++k) {
    int d = k * 256 + tid;
    x[k] = qx[(size_t)row * Dd + d] - mean[d];
    ss += x[k] * x[k];
  }
  red[tid] = ss; __syncthreads();
  for (int s = 128; s > 0; s >>= 1) { if (tid < s) red[tid] += red[tid + s]; __syncthreads(); }
  float rn = rsqrtf(red[0]);
#pragma unroll
  for (int k = 0; k < 4; ++k) {
    int d = k * 256 + tid;
    qn[(size_t)row * Dd + d] = f2bf(x[k] * rn);
  }
}

// ---------------------------------------------------------------- LDS-tiled transpose qn -> qnT (64x64 bf16 tiles)
__global__ __launch_bounds__(256)
void transpose_k(const u16* __restrict__ qn, u16* __restrict__ qnT) {
  __shared__ u16 tile[64][72];              // row stride 144 B (16B multiple), pad kills bank conflicts
  const int qb = blockIdx.x * 64, db = blockIdx.y * 64;
  {
    int r = threadIdx.x >> 2;               // query row within tile
    int cseg = (threadIdx.x & 3) * 16;      // d segment
    const u16* src = qn + (size_t)(qb + r) * Dd + db + cseg;
    int4 v0 = *(const int4*)src;
    int4 v1 = *(const int4*)(src + 8);
    *(int4*)&tile[r][cseg]     = v0;
    *(int4*)&tile[r][cseg + 8] = v1;
  }
  __syncthreads();
  {
    int dr = threadIdx.x >> 2;              // d row within tile
    int qseg = (threadIdx.x & 3) * 16;      // query segment
    Pack8 p0, p1;
#pragma unroll
    for (int i = 0; i < 8; ++i) p0.h[i] = tile[qseg + i][dr];
#pragma unroll
    for (int i = 0; i < 8; ++i) p1.h[i] = tile[qseg + 8 + i][dr];
    u16* dst = qnT + (size_t)(db + dr) * NQs + qb + qseg;
    *(int4*)dst       = p0.q;
    *(int4*)(dst + 8) = p1.q;
  }
}

// ---------------------------------------------------------------- centroid update: reduce split-K partials, cent_bf + ||c||^2
__global__ __launch_bounds__(256)
void update_k(const float* __restrict__ sup_sum, const float* __restrict__ sup_cnt,
              const float* __restrict__ wqp, const float* __restrict__ wsum,
              u16* __restrict__ cent_bf, float* __restrict__ cnorm2) {
  int c = blockIdx.x, tid = threadIdx.x;
  __shared__ float red[256];
  float denom = fmaxf(sup_cnt[c] + wsum[c], 1e-6f);
  float ss = 0.f;
#pragma unroll
  for (int k = 0; k < 4; ++k) {
    int d = k * 256 + tid;
    float acc = sup_sum[(size_t)c * Dd + d];
    for (int kk = 0; kk < KSPLIT; ++kk)
      acc += wqp[(size_t)kk * CP * Dd + (size_t)c * Dd + d];
    float v = (c < Cc) ? acc / denom : 0.f;
    u16 b = f2bf(v);
    cent_bf[(size_t)c * Dd + d] = b;
    float vb = bf2f(b);
    ss += vb * vb;
  }
  red[tid] = ss; __syncthreads();
  for (int s = 128; s > 0; s >>= 1) { if (tid < s) red[tid] += red[tid + s]; __syncthreads(); }
  if (tid == 0) cnorm2[c] = red[0];
}

// ---------------------------------------------------------------- GEMM1 + fused softmax / argmin
// 8 waves/block, 128 query rows/block; each wave: 16 rows x 112 classes (7 tiles).
__global__ __launch_bounds__(256)
void assign_k(const u16* __restrict__ qn, const u16* __restrict__ cent_bf,
              const float* __restrict__ cnorm2, const int* __restrict__ beta_p,
              const int* __restrict__ qy, u16* __restrict__ wT,
              float* __restrict__ wsum, int* __restrict__ correct, int pred_mode) {
  const int lane = threadIdx.x & 31;
  const int wv   = threadIdx.x >> 5;
  const int m0   = blockIdx.x * 128 + wv * 16;
  const bool hi  = lane >= 16;
  const int  ln  = lane & 15;

  v8f acc[7];
#pragma unroll
  for (int t = 0; t < 7; ++t) acc[t] = (v8f)0.0f;

  const u16* arow = qn + (size_t)(m0 + ln) * Dd + (hi ? 8 : 0);
  for (int k0 = 0; k0 < Dd; k0 += 32) {
    BF16Frag a;                               // A 16x32: lane=M; K chunks {0..7,16..23} / {8..15,24..31}
    a.q[0] = *(const int4*)(arow + k0);
    a.q[1] = *(const int4*)(arow + k0 + 16);
#pragma unroll
    for (int t = 0; t < 7; ++t) {
      BF16Frag b;                             // B 32x16: lane=N; K chunks {0..7,16..23} / {8..15,24..31}
      const u16* bp = cent_bf + (size_t)(t * 16 + ln) * Dd + k0 + (hi ? 8 : 0);
      b.q[0] = *(const int4*)bp;
      b.q[1] = *(const int4*)(bp + 16);
      acc[t] = __builtin_amdgcn_wmma_f32_16x16x32_bf16(
                 false, a.v, false, b.v, (short)0, acc[t], false, false);
    }
  }

  const float beta = (float)(*beta_p);

  if (!pred_mode) {
    float w[7][8], mrow[8], srow[8], inv[8];
#pragma unroll
    for (int j = 0; j < 8; ++j) mrow[j] = -1e30f;
#pragma unroll
    for (int t = 0; t < 7; ++t) {
      float c2 = cnorm2[t * 16 + ln];
      bool valid = (t * 16 + ln) < Cc;
#pragma unroll
      for (int j = 0; j < 8; ++j) {
        float d2 = 1.0f + c2 - 2.0f * acc[t][j];
        float lg = valid ? -beta * d2 : -1e30f;
        w[t][j] = lg;
        mrow[j] = fmaxf(mrow[j], lg);
      }
    }
#pragma unroll
    for (int m = 1; m < 16; m <<= 1)
#pragma unroll
      for (int j = 0; j < 8; ++j) mrow[j] = fmaxf(mrow[j], __shfl_xor(mrow[j], m, 32));
#pragma unroll
    for (int j = 0; j < 8; ++j) srow[j] = 0.f;
#pragma unroll
    for (int t = 0; t < 7; ++t)
#pragma unroll
      for (int j = 0; j < 8; ++j) { float e = __expf(w[t][j] - mrow[j]); w[t][j] = e; srow[j] += e; }
#pragma unroll
    for (int m = 1; m < 16; m <<= 1)
#pragma unroll
      for (int j = 0; j < 8; ++j) srow[j] += __shfl_xor(srow[j], m, 32);
#pragma unroll
    for (int j = 0; j < 8; ++j) inv[j] = 1.0f / srow[j];

    float p[7];
#pragma unroll
    for (int t = 0; t < 7; ++t) {
      int cls = t * 16 + ln;
      p[t] = 0.f;
      Pack8 pk;
#pragma unroll
      for (int j = 0; j < 8; ++j) {
        float wv_ = w[t][j] * inv[j];
        p[t] += wv_;
        pk.h[j] = f2bf(wv_);
      }
      // rows m0..m0+7 (lo half) / m0+8..m0+15 (hi half): one 16B store per class tile
      *(int4*)(wT + (size_t)cls * NQs + m0 + (hi ? 8 : 0)) = pk.q;
    }
#pragma unroll
    for (int t = 0; t < 7; ++t) p[t] += __shfl_xor(p[t], 16, 32);
    if (!hi) {
#pragma unroll
      for (int t = 0; t < 7; ++t) atomicAdd(&wsum[t * 16 + ln], p[t]);
    }
  } else {
    float bd[8]; int bi[8];
#pragma unroll
    for (int j = 0; j < 8; ++j) { bd[j] = 1e30f; bi[j] = 0x7fffffff; }
#pragma unroll
    for (int t = 0; t < 7; ++t) {
      float c2 = cnorm2[t * 16 + ln];
      int cls = t * 16 + ln;
      bool valid = cls < Cc;
#pragma unroll
      for (int j = 0; j < 8; ++j) {
        float d2 = valid ? (1.0f + c2 - 2.0f * acc[t][j]) : 1e30f;
        if (d2 < bd[j]) { bd[j] = d2; bi[j] = cls; }
      }
    }
#pragma unroll
    for (int m = 1; m < 16; m <<= 1) {
#pragma unroll
      for (int j = 0; j < 8; ++j) {
        float od = __shfl_xor(bd[j], m, 32);
        int   oi = __shfl_xor(bi[j], m, 32);
        if (od < bd[j]) { bd[j] = od; bi[j] = oi; }
      }
    }
    if (ln == 0) {
      int cnt = 0;
#pragma unroll
      for (int j = 0; j < 8; ++j) {
        int row = m0 + j + (hi ? 8 : 0);
        cnt += (bi[j] == qy[row]) ? 1 : 0;
      }
      atomicAdd(correct, cnt);
    }
  }
}

// ---------------------------------------------------------------- GEMM2: wqp[ks][C][D] = w^T * q over 1024-query slice
// 4 waves/block; wave owns one 16-col D tile x all 7 class tiles; plain stores (no atomics).
__global__ __launch_bounds__(128)
void gemm2_k(const u16* __restrict__ wT, const u16* __restrict__ qnT,
             float* __restrict__ wqp) {
  const int lane = threadIdx.x & 31;
  const int wv   = threadIdx.x >> 5;
  const bool hi  = lane >= 16;
  const int  ln  = lane & 15;
  const int d0    = blockIdx.x * 64 + wv * 16;
  const int qbase = blockIdx.y * 1024;

  v8f acc[7];
#pragma unroll
  for (int t = 0; t < 7; ++t) acc[t] = (v8f)0.0f;

  const u16* brow = qnT + (size_t)(d0 + ln) * NQs + (hi ? 8 : 0);
  for (int ks = 0; ks < 32; ++ks) {
    int q0 = qbase + ks * 32;
    BF16Frag b;                               // B 32x16: lane=N(d); K=queries chunks per half
    b.q[0] = *(const int4*)(brow + q0);
    b.q[1] = *(const int4*)(brow + q0 + 16);
#pragma unroll
    for (int t = 0; t < 7; ++t) {
      BF16Frag a;                             // A 16x32: lane=M(class); K chunks per half
      const u16* ap = wT + (size_t)(t * 16 + ln) * NQs + q0 + (hi ? 8 : 0);
      a.q[0] = *(const int4*)ap;
      a.q[1] = *(const int4*)(ap + 16);
      acc[t] = __builtin_amdgcn_wmma_f32_16x16x32_bf16(
                 false, a.v, false, b.v, (short)0, acc[t], false, false);
    }
  }
  float* out = wqp + (size_t)blockIdx.y * CP * Dd;
#pragma unroll
  for (int t = 0; t < 7; ++t)
#pragma unroll
    for (int j = 0; j < 8; ++j) {
      int cls = t * 16 + j + (hi ? 8 : 0);
      int d   = d0 + ln;
      out[(size_t)cls * Dd + d] = acc[t][j];
    }
}

// ---------------------------------------------------------------- finalize
__global__ void fin_k(const int* __restrict__ correct, float* __restrict__ out) {
  if (threadIdx.x == 0 && blockIdx.x == 0)
    out[0] = (float)(*correct) * (1.0f / (float)NQs);
}

// ---------------------------------------------------------------- launcher
extern "C" void kernel_launch(void* const* d_in, const int* in_sizes, int n_in,
                              void* d_out, int out_size, void* d_ws, size_t ws_size,
                              hipStream_t stream) {
  (void)in_sizes; (void)n_in; (void)out_size; (void)ws_size;
  const float* sx = (const float*)d_in[0];
  const int*   sy = (const int*)  d_in[1];
  const float* qx = (const float*)d_in[2];
  const int*   qy = (const int*)  d_in[3];
  const int*   beta_p = (const int*)d_in[5];

  char* p = (char*)d_ws;
  u16*  qn      = (u16*)p;  p += (size_t)NQs * Dd * 2;          // 256 MB
  u16*  qnT     = (u16*)p;  p += (size_t)Dd * NQs * 2;          // 256 MB
  u16*  wT      = (u16*)p;  p += (size_t)CP * NQs * 2;          // 28 MB
  u16*  cent_bf = (u16*)p;  p += (size_t)CP * Dd * 2;
  float* mean    = (float*)p; p += (size_t)Dd * 4;
  float* sup_sum = (float*)p; p += (size_t)CP * Dd * 4;
  float* sup_cnt = (float*)p; p += 256;
  float* wqp     = (float*)p; p += (size_t)KSPLIT * CP * Dd * 4; // 58.7 MB split-K partials
  float* wsum    = (float*)p; p += 256;
  float* cnorm2  = (float*)p; p += 256;
  int*   correct = (int*)p;   p += 256;

  zero_sup_k<<<(CP * Dd + 255) / 256, 256, 0, stream>>>(sup_sum, sup_cnt);
  zero_wqp_k<<<(KSPLIT * CP * Dd + 255) / 256, 256, 0, stream>>>(wqp);
  mean_k<<<Dd / 256, 256, 0, stream>>>(sx, mean);
  sup_k<<<NSs, 256, 0, stream>>>(sx, sy, mean, sup_sum, sup_cnt);
  qnorm_k<<<NQs, 256, 0, stream>>>(qx, mean, qn);
  transpose_k<<<dim3(NQs / 64, Dd / 64), 256, 0, stream>>>(qn, qnT);
  zero_iter_k<<<1, 256, 0, stream>>>(wsum, correct);
  update_k<<<CP, 256, 0, stream>>>(sup_sum, sup_cnt, wqp, wsum, cent_bf, cnorm2); // centroids0

  for (int it = 0; it < NIT; ++it) {
    assign_k<<<NQs / 128, 256, 0, stream>>>(qn, cent_bf, cnorm2, beta_p, qy,
                                            wT, wsum, correct, /*pred=*/0);
    gemm2_k<<<dim3(Dd / 64, NQs / 1024), 128, 0, stream>>>(wT, qnT, wqp);
    update_k<<<CP, 256, 0, stream>>>(sup_sum, sup_cnt, wqp, wsum, cent_bf, cnorm2);
    zero_iter_k<<<1, 256, 0, stream>>>(wsum, correct);
  }

  assign_k<<<NQs / 128, 256, 0, stream>>>(qn, cent_bf, cnorm2, beta_p, qy,
                                          wT, wsum, correct, /*pred=*/1);
  fin_k<<<1, 64, 0, stream>>>(correct, (float*)d_out);
}
